// PPO_72739566125955
// MI455X (gfx1250) — matrix operand order repaired and checked by reading, not compiled, not run on hardware
//
#include <hip/hip_runtime.h>

// ---------------------------------------------------------------------------
// MI455X (gfx1250) fused retrieval-PPO forward.
// Heavy math via v_wmma_f32_16x16x32_f16 with register double-buffered
// B tiles (loads overlap WMMA), wave-private LDS bounce for top-k (no block
// barriers in the hot loop), M split in 2 for grid occupancy.
// ---------------------------------------------------------------------------

typedef _Float16 v16h __attribute__((ext_vector_type(16)));
typedef _Float16 v8h  __attribute__((ext_vector_type(8)));
typedef float    v8f  __attribute__((ext_vector_type(8)));

constexpr int B    = 4096;
constexpr int M    = 32768;
constexpr int S    = 256;
constexpr int H    = 1024;
constexpr int TOPK = 4;
constexpr int S2   = 2 * S;
constexpr int MPARTS = 2;
constexpr int TILES  = (M / MPARTS) / 128;  // 128 tiles of 128 columns

#define NEG_BIG (-3.402823466e38f)

__device__ __forceinline__ v8f wmma_f16(v16h a, v16h b, v8f c) {
  return __builtin_amdgcn_wmma_f32_16x16x32_f16(
      false, a, false, b, (short)0, c, false, false);
}

// A-matrix 16x32 f16 fragment (M x K), contiguous K order (matches B):
//   lanes 0-15 : row = lane,    K = {k0..k0+15}
//   lanes 16-31: row = lane-16, K = {k0+16..k0+31}
__device__ __forceinline__ v16h load_a_frag(const _Float16* __restrict__ rowp,
                                            int k0, int hi) {
  const _Float16* p = rowp + k0 + 16 * hi;
  v8h lo = *(const v8h*)(p);
  v8h hh = *(const v8h*)(p + 8);
  v16h a;
#pragma unroll
  for (int i = 0; i < 8; ++i) { a[i] = lo[i]; a[i + 8] = hh[i]; }
  return a;
}

// B-matrix 32x16 fragment tile: one column's 16 contiguous K values per lane.
// Loads a full K=256 strip (8 k-steps) for this lane's column.
__device__ __forceinline__ void load_btile(v16h (&bf)[8],
                                           const _Float16* __restrict__ base,
                                           int hi) {
#pragma unroll
  for (int kk = 0; kk < 8; ++kk)
    bf[kk] = *(const v16h*)(base + kk * 32 + 16 * hi);
}

// top-4 update from 16 candidates (increasing index order, strict > = stable)
__device__ __forceinline__ void topk_insert16(const float (&vals)[16], int m0,
                                              float (&tv)[4], int (&ti)[4]) {
  float mx = vals[0];
#pragma unroll
  for (int i = 1; i < 16; ++i) mx = fmaxf(mx, vals[i]);
  if (mx > tv[3]) {
#pragma unroll
    for (int c2 = 0; c2 < 16; ++c2) {
      float v = vals[c2];
      if (v > tv[3]) {
        tv[3] = v; ti[3] = m0 + c2;
#pragma unroll
        for (int q = 3; q > 0; --q)
          if (tv[q] > tv[q - 1]) {
            float fv = tv[q]; tv[q] = tv[q - 1]; tv[q - 1] = fv;
            int   fi = ti[q]; ti[q] = ti[q - 1]; ti[q - 1] = fi;
          }
      }
    }
  }
}

// ---------------------------------------------------------------------------
// f32 -> f16 conversion
// ---------------------------------------------------------------------------
__global__ void cvt_f32_f16_kernel(const float* __restrict__ in,
                                   _Float16* __restrict__ out, int n) {
  for (int i = blockIdx.x * blockDim.x + threadIdx.x; i < n;
       i += gridDim.x * blockDim.x)
    out[i] = (_Float16)in[i];
}

// ---------------------------------------------------------------------------
// Fused similarity GEMM + running top-4.
// Block = 32 x-rows, half of M (blockIdx.y). 8 waves, each owns a 16-col
// slice of every 128-wide tile. A resident; B register double-buffered.
// ---------------------------------------------------------------------------
__device__ __forceinline__ void sim_tile_step(
    const v16h (&a0)[8], const v16h (&a1)[8],
    const v16h (&curb)[8], v16h (&nxtb)[8],
    const _Float16* __restrict__ nxt_base, bool pf,
    int hi, int col, int lane, float (*st)[16], int m0,
    float (&tv)[4], int (&ti)[4]) {
  if (pf) {
    load_btile(nxtb, nxt_base, hi);                       // tile t+1 in flight
    __builtin_prefetch(nxt_base + (size_t)128 * S, 0, 0); // tile t+2 into L0
  }
  v8f c0 = {}, c1 = {};
#pragma unroll
  for (int kk = 0; kk < 8; ++kk) {
    c0 = wmma_f16(a0[kk], curb[kk], c0);
    c1 = wmma_f16(a1[kk], curb[kk], c1);
  }
  // C 16x16: lane holds col, rows r + 8*hi (group0: rows 0-15, group1: 16-31)
#pragma unroll
  for (int r = 0; r < 8; ++r) {
    st[r + 8 * hi][col]      = c0[r];
    st[16 + r + 8 * hi][col] = c1[r];
  }
  __builtin_amdgcn_wave_barrier();  // wave-private LDS: scheduling fence only
  const float4* rp = (const float4*)&st[lane][0];
  float4 q0 = rp[0], q1 = rp[1], q2 = rp[2], q3 = rp[3];
  const float vals[16] = {q0.x, q0.y, q0.z, q0.w, q1.x, q1.y, q1.z, q1.w,
                          q2.x, q2.y, q2.z, q2.w, q3.x, q3.y, q3.z, q3.w};
  topk_insert16(vals, m0, tv, ti);
  __builtin_amdgcn_wave_barrier();  // reads done before next tile overwrites
}

__global__ __launch_bounds__(256) void sim_topk_kernel(
    const _Float16* __restrict__ xh, const _Float16* __restrict__ kh,
    float* __restrict__ part_v, int* __restrict__ part_i) {
  __shared__ __align__(16) float stile[8][32][16];
  __shared__ float wtv[8][32][TOPK];
  __shared__ int   wti[8][32][TOPK];

  const int tid  = threadIdx.x;
  const int wave = tid >> 5;
  const int lane = tid & 31;
  const int col  = lane & 15;
  const int hi   = lane >> 4;
  const int brow0   = blockIdx.x * 32;
  const int part    = blockIdx.y;
  const int m_part0 = part * (M / MPARTS);

  // resident A fragments: 32 rows x K=256 (two 16-row groups)
  v16h a0[8], a1[8];
  const _Float16* ar0 = xh + (size_t)(brow0 + col) * S;
  const _Float16* ar1 = xh + (size_t)(brow0 + 16 + col) * S;
#pragma unroll
  for (int kk = 0; kk < 8; ++kk) {
    a0[kk] = load_a_frag(ar0, kk * 32, hi);
    a1[kk] = load_a_frag(ar1, kk * 32, hi);
  }

  float tv[TOPK];
  int   ti[TOPK];
#pragma unroll
  for (int k = 0; k < TOPK; ++k) { tv[k] = NEG_BIG; ti[k] = 0x7FFFFFFF; }

  const _Float16* bbase = kh + (size_t)(m_part0 + wave * 16 + col) * S;
  float (*st)[16] = stile[wave];

  v16h b0[8], b1[8];
  load_btile(b0, bbase, hi);

  for (int tile = 0; tile < TILES; tile += 2) {  // TILES even; idx compile-time
    sim_tile_step(a0, a1, b0, b1, bbase + (size_t)(tile + 1) * 128 * S,
                  tile + 1 < TILES, hi, col, lane, st,
                  m_part0 + tile * 128 + wave * 16, tv, ti);
    sim_tile_step(a0, a1, b1, b0, bbase + (size_t)(tile + 2) * 128 * S,
                  tile + 2 < TILES, hi, col, lane, st,
                  m_part0 + (tile + 1) * 128 + wave * 16, tv, ti);
  }

#pragma unroll
  for (int k = 0; k < TOPK; ++k) {
    wtv[wave][lane][k] = tv[k];
    wti[wave][lane][k] = ti[k];
  }
  __syncthreads();

  // merge 8 per-wave top-4 lists: value desc, index asc on ties
  if (tid < 32) {
    const int row = tid;
    unsigned taken = 0;
    for (int k = 0; k < TOPK; ++k) {
      float bv = NEG_BIG;
      int   bi = 0x7FFFFFFF;
      int   bs = 0;
      for (int w = 0; w < 8; ++w)
        for (int q = 0; q < TOPK; ++q) {
          int s = w * TOPK + q;
          if (taken & (1u << s)) continue;
          float v = wtv[w][row][q];
          int   i = wti[w][row][q];
          if (v > bv || (v == bv && i < bi)) { bv = v; bi = i; bs = s; }
        }
      taken |= (1u << bs);
      const size_t o = ((size_t)part * B + brow0 + row) * TOPK + k;
      part_v[o] = bv;
      part_i[o] = bi;
    }
  }
}

// ---------------------------------------------------------------------------
// Merge the two M-halves' top-4 lists (part 0 has the smaller indices).
// ---------------------------------------------------------------------------
__global__ void topk_merge_kernel(const float* __restrict__ pv,
                                  const int* __restrict__ pi,
                                  int* __restrict__ kidx) {
  const int b = blockIdx.x * blockDim.x + threadIdx.x;
  if (b >= B) return;
  float v[2 * TOPK];
  int   ix[2 * TOPK];
#pragma unroll
  for (int p = 0; p < 2; ++p)
#pragma unroll
    for (int k = 0; k < TOPK; ++k) {
      const size_t o = ((size_t)p * B + b) * TOPK + k;
      v[p * TOPK + k]  = pv[o];
      ix[p * TOPK + k] = pi[o];
    }
  unsigned taken = 0;
#pragma unroll
  for (int k = 0; k < TOPK; ++k) {
    float bv = NEG_BIG;
    int   bi = 0x7FFFFFFF;
    int   bs = 0;
#pragma unroll
    for (int s = 0; s < 2 * TOPK; ++s) {
      if (taken & (1u << s)) continue;
      if (v[s] > bv || (v[s] == bv && ix[s] < bi)) { bv = v[s]; bi = ix[s]; bs = s; }
    }
    taken |= (1u << bs);
    kidx[(size_t)b * TOPK + k] = bi;
  }
}

// ---------------------------------------------------------------------------
// Gather v_hat = values_mem[k_idx] -> f16
// ---------------------------------------------------------------------------
__global__ void gather_vhat_kernel(const float* __restrict__ vals,
                                   const int* __restrict__ kidx,
                                   _Float16* __restrict__ vh) {
  const int row = blockIdx.x;
  const int idx = kidx[row];
  vh[(size_t)row * S + threadIdx.x] =
      (_Float16)vals[(size_t)idx * S + threadIdx.x];
}

// ---------------------------------------------------------------------------
// Value net: pv = Wv . relu(W1 v_hat + b1) + bv, then rank-of-slot-0 select.
// B tiles register double-buffered across the 8 H-iterations.
// ---------------------------------------------------------------------------
__device__ __forceinline__ void vnet_step(
    const v16h (&af)[8], const v16h (&curb)[8], v16h (&nxtb)[8],
    const _Float16* __restrict__ nxt_base, bool pf, int hi, int h,
    const float* __restrict__ b1, const float* __restrict__ Wv,
    float (&acc)[8]) {
  if (pf) load_btile(nxtb, nxt_base, hi);
  v8f c = {};
#pragma unroll
  for (int kk = 0; kk < 8; ++kk) c = wmma_f16(af[kk], curb[kk], c);
  const float bias = b1[h], wv = Wv[h];
#pragma unroll
  for (int r = 0; r < 8; ++r) acc[r] += fmaxf(c[r] + bias, 0.f) * wv;
}

__global__ __launch_bounds__(256) void vnet_select_kernel(
    const _Float16* __restrict__ vh, const _Float16* __restrict__ w1h,
    const float* __restrict__ b1, const float* __restrict__ Wv,
    const float* __restrict__ bvp, const int* __restrict__ kidx,
    int* __restrict__ sel) {
  __shared__ float pv[16];

  const int tid  = threadIdx.x;
  const int wave = tid >> 5;
  const int lane = tid & 31;
  const int col  = lane & 15;
  const int hi   = lane >> 4;
  const int r0   = blockIdx.x * 16;  // rows in (b, t) space

  v16h afrag[8];
  const _Float16* arow = vh + (size_t)(r0 + col) * S;
#pragma unroll
  for (int kk = 0; kk < 8; ++kk) afrag[kk] = load_a_frag(arow, kk * 32, hi);

  float acc[8];
#pragma unroll
  for (int r = 0; r < 8; ++r) acc[r] = 0.f;

  const _Float16* wbase = w1h + (size_t)(wave * 16 + col) * S;
  v16h bb0[8], bb1[8];
  load_btile(bb0, wbase, hi);

  for (int it = 0; it < H / 128; it += 2) {
    vnet_step(afrag, bb0, bb1, wbase + (size_t)(it + 1) * 128 * S,
              it + 1 < H / 128, hi, it * 128 + wave * 16 + col, b1, Wv, acc);
    vnet_step(afrag, bb1, bb0, wbase + (size_t)(it + 2) * 128 * S,
              it + 2 < H / 128, hi, (it + 1) * 128 + wave * 16 + col, b1, Wv,
              acc);
  }

  if (tid < 16) pv[tid] = bvp[0];
  __syncthreads();
#pragma unroll
  for (int r = 0; r < 8; ++r) atomicAdd(&pv[r + 8 * hi], acc[r]);
  __syncthreads();

  if (tid < 4) {
    const int base = 4 * tid;
    const float p0 = pv[base];
    // rank of element 0 under stable ascending argsort
    const int j = (pv[base + 1] < p0) + (pv[base + 2] < p0) + (pv[base + 3] < p0);
    const int bglob = r0 / TOPK + tid;
    sel[bglob] = kidx[(size_t)bglob * TOPK + j];
  }
}

// ---------------------------------------------------------------------------
// cat = [x_f16 | f16(values_mem[sel])]
// ---------------------------------------------------------------------------
__global__ void build_cat_kernel(const _Float16* __restrict__ xh,
                                 const float* __restrict__ vals,
                                 const int* __restrict__ sel,
                                 _Float16* __restrict__ cath) {
  const int b = blockIdx.x;
  const int t = threadIdx.x;  // 512 threads
  _Float16 v = (t < S) ? xh[(size_t)b * S + t]
                       : (_Float16)vals[(size_t)sel[b] * S + (t - S)];
  cath[(size_t)b * S2 + t] = v;
}

// ---------------------------------------------------------------------------
// Head: logits = Wpi relu(W2 cat + b2) + bpi ; softmax over 2.
// Per H-tile: batch all 32 b128 loads, then 16 back-to-back WMMAs.
// ---------------------------------------------------------------------------
__global__ __launch_bounds__(256) void head_kernel(
    const _Float16* __restrict__ cath, const _Float16* __restrict__ w2h,
    const float* __restrict__ b2, const float* __restrict__ Wpi,
    const float* __restrict__ bpi, float* __restrict__ out) {
  __shared__ float sl[16][2];

  const int tid  = threadIdx.x;
  const int wave = tid >> 5;
  const int lane = tid & 31;
  const int col  = lane & 15;
  const int hi   = lane >> 4;
  const int brow0 = blockIdx.x * 16;

  v16h afrag[16];  // 16 rows x K=512 resident
  const _Float16* arow = cath + (size_t)(brow0 + col) * S2;
#pragma unroll
  for (int kk = 0; kk < 16; ++kk) afrag[kk] = load_a_frag(arow, kk * 32, hi);

  float a0[8] = {}, a1[8] = {};

  for (int it = 0; it < H / 128; ++it) {
    const int h0 = it * 128 + wave * 16;
    const _Float16* bcol = w2h + (size_t)(h0 + col) * S2;
    v16h bf[16];
#pragma unroll
    for (int kk = 0; kk < 16; ++kk)
      bf[kk] = *(const v16h*)(bcol + kk * 32 + 16 * hi);
    v8f c = {};
#pragma unroll
    for (int kk = 0; kk < 16; ++kk) c = wmma_f16(afrag[kk], bf[kk], c);

    const int h = h0 + col;
    const float bias = b2[h];
    const float w0 = Wpi[h];
    const float w1 = Wpi[H + h];
#pragma unroll
    for (int r = 0; r < 8; ++r) {
      float v = fmaxf(c[r] + bias, 0.f);
      a0[r] += v * w0;
      a1[r] += v * w1;
    }
  }

  if (tid < 16) { sl[tid][0] = bpi[0]; sl[tid][1] = bpi[1]; }
  __syncthreads();
#pragma unroll
  for (int r = 0; r < 8; ++r) {
    atomicAdd(&sl[r + 8 * hi][0], a0[r]);
    atomicAdd(&sl[r + 8 * hi][1], a1[r]);
  }
  __syncthreads();

  if (tid < 16) {
    float l0 = sl[tid][0], l1 = sl[tid][1];
    float mx = fmaxf(l0, l1);
    float e0 = __expf(l0 - mx), e1 = __expf(l1 - mx);
    float inv = 1.f / (e0 + e1);
    out[(size_t)(brow0 + tid) * 2 + 0] = e0 * inv;
    out[(size_t)(brow0 + tid) * 2 + 1] = e1 * inv;
  }
}

// ---------------------------------------------------------------------------
// Host launch
// ---------------------------------------------------------------------------
extern "C" void kernel_launch(void* const* d_in, const int* in_sizes, int n_in,
                              void* d_out, int out_size, void* d_ws,
                              size_t ws_size, hipStream_t stream) {
  (void)in_sizes; (void)n_in; (void)out_size; (void)ws_size;

  const float* x    = (const float*)d_in[0];
  const float* keys = (const float*)d_in[1];
  const float* vals = (const float*)d_in[2];
  const float* W1   = (const float*)d_in[3];
  const float* b1   = (const float*)d_in[4];
  const float* Wv   = (const float*)d_in[5];
  const float* bv   = (const float*)d_in[6];
  const float* W2   = (const float*)d_in[7];
  const float* b2   = (const float*)d_in[8];
  const float* Wpi  = (const float*)d_in[9];
  const float* bpi  = (const float*)d_in[10];
  float* out = (float*)d_out;

  // workspace layout (all offsets well aligned)
  char* ws = (char*)d_ws;
  size_t off = 0;
  _Float16* x_h    = (_Float16*)(ws + off); off += (size_t)B * S * 2;
  _Float16* keys_h = (_Float16*)(ws + off); off += (size_t)M * S * 2;
  _Float16* W1_h   = (_Float16*)(ws + off); off += (size_t)H * S * 2;
  _Float16* W2_h   = (_Float16*)(ws + off); off += (size_t)H * S2 * 2;
  _Float16* vhat_h = (_Float16*)(ws + off); off += (size_t)B * TOPK * S * 2;
  _Float16* cat_h  = (_Float16*)(ws + off); off += (size_t)B * S2 * 2;
  float* part_v    = (float*)(ws + off);    off += (size_t)MPARTS * B * TOPK * 4;
  int* part_i      = (int*)(ws + off);      off += (size_t)MPARTS * B * TOPK * 4;
  int* kidx        = (int*)(ws + off);      off += (size_t)B * TOPK * 4;
  int* sel         = (int*)(ws + off);      off += (size_t)B * 4;

  // 1) f32 -> f16 conversions
  cvt_f32_f16_kernel<<<1024, 256, 0, stream>>>(x, x_h, B * S);
  cvt_f32_f16_kernel<<<2048, 256, 0, stream>>>(keys, keys_h, M * S);
  cvt_f32_f16_kernel<<<512, 256, 0, stream>>>(W1, W1_h, H * S);
  cvt_f32_f16_kernel<<<512, 256, 0, stream>>>(W2, W2_h, H * S2);

  // 2) fused similarity GEMM + top-4 over each M half
  sim_topk_kernel<<<dim3(B / 32, MPARTS), 256, 0, stream>>>(x_h, keys_h,
                                                            part_v, part_i);
  // 2b) merge halves
  topk_merge_kernel<<<B / 256, 256, 0, stream>>>(part_v, part_i, kidx);

  // 3) gather retrieved value vectors
  gather_vhat_kernel<<<B * TOPK, S, 0, stream>>>(vals, kidx, vhat_h);

  // 4) value net + rank-of-slot-0 selection
  vnet_select_kernel<<<(B * TOPK) / 16, 256, 0, stream>>>(vhat_h, W1_h, b1, Wv,
                                                          bv, kidx, sel);

  // 5) build concat input
  build_cat_kernel<<<B, S2, 0, stream>>>(x_h, vals, sel, cat_h);

  // 6) policy head + softmax
  head_kernel<<<B / 16, 256, 0, stream>>>(cat_h, W2_h, b2, Wpi, bpi, out);
}
